// CKConv_22497038697104
// MI455X (gfx1250) — matrix operand order repaired and checked by reading, not compiled, not run on hardware
//
#include <hip/hip_runtime.h>
#include <cmath>

// CKConv on MI455X (gfx1250, wave32):
//   Phase A: SIREN kernel generator -> f16 kernel in A-operand layout Akern[co][j][ci]
//   Phase B: causal conv as Toeplitz implicit GEMM (M=64 co, N=B*L, K=j*ci=131072)
//            using v_wmma_f32_16x16x32_f16, split-K over j with f32 atomics.
//            x-band staged into LDS with GLOBAL_LOAD_ASYNC_TO_LDS_B128 (ASYNCcnt).

typedef _Float16 v8h  __attribute__((ext_vector_type(8)));
typedef _Float16 v16h __attribute__((ext_vector_type(16)));
typedef float    v8f  __attribute__((ext_vector_type(8)));

#define OMEGA0 30.0f
#define LEN    2048
#define CIN    64
#define COUT   64
#define HID    128
#define NB     2

static __device__ inline v16h cat8(v8h lo, v8h hi) {
  return __builtin_shufflevector(lo, hi, 0,1,2,3,4,5,6,7,8,9,10,11,12,13,14,15);
}

#if __has_builtin(__builtin_amdgcn_s_wait_asynccnt)
#define WAIT_ASYNC0() __builtin_amdgcn_s_wait_asynccnt(0)
#else
#define WAIT_ASYNC0() asm volatile("s_wait_asynccnt 0x0" ::: "memory")
#endif

// ---- K0: x (B,Cin,L) f32 -> xh (B,L,Cin) f16 (time-major, channel-contiguous) ----
__global__ void ck_cvt_x(const float* __restrict__ x, _Float16* __restrict__ xh) {
  const int i  = blockIdx.x * 256 + threadIdx.x;      // grid sized exactly: B*L*Cin
  const int ci = i & 63;
  const int s  = (i >> 6) & (LEN - 1);
  const int b  = i >> 17;
  xh[i] = (_Float16)x[(b << 17) + (ci << 11) + s];
}

// ---- K0b: W3 (128,4096) f32 -> W3t (4096,128) f16 (B-operand: N-major, K-contiguous) ----
__global__ void ck_cvt_w3(const float* __restrict__ W3, _Float16* __restrict__ W3t) {
  const int i = blockIdx.x * 256 + threadIdx.x;       // grid sized exactly: 4096*128
  const int k = i & 127;
  const int n = i >> 7;
  W3t[i] = (_Float16)W3[(k << 12) + n];
}

// ---- K1: SIREN layers 1+2 in f32 VALU: h2[l][t] = sin(30*(sin(30*rel*W1) @ W2)) ----
__global__ void __launch_bounds__(128) ck_siren(const float* __restrict__ W1,
                                                const float* __restrict__ W2,
                                                _Float16* __restrict__ h2h) {
  const int l = blockIdx.x;          // 0..2047
  const int t = threadIdx.x;         // 0..127
  __shared__ float h1[HID];
  const float rel = -1.0f + 2.0f * (float)l / (float)(LEN - 1);
  h1[t] = sinf(OMEGA0 * rel * W1[t]);
  __syncthreads();
  float acc = 0.0f;
#pragma unroll 8
  for (int k = 0; k < HID; ++k)
    acc = fmaf(h1[k], W2[(k << 7) + t], acc);          // coalesced across threads
  h2h[(l << 7) + t] = (_Float16)sinf(OMEGA0 * acc);
}

// ---- K2: kgen = h2(2048x128) @ W3(128x4096) via WMMA; write Akern[co][j][ci] f16 ----
__global__ void __launch_bounds__(256) ck_gen_gemm(const _Float16* __restrict__ h2h,
                                                   const _Float16* __restrict__ W3t,
                                                   _Float16* __restrict__ Akern) {
  const int tid  = threadIdx.x;
  const int wave = tid >> 5, lane = tid & 31;
  const int r = lane & 15, hi = lane >> 4;
  const int gw    = blockIdx.x * 8 + wave;             // 32768 subtiles total
  const int mtile = gw >> 8;                           // 0..127 (j rows)
  const int ntile = gw & 255;                          // 0..255 (co*64+ci cols)
  const int j0 = mtile << 4, n0 = ntile << 4;

  const _Float16* pA = h2h + ((j0 + r) << 7) + (hi << 3);   // row j0+r, K base hi*8
  const _Float16* pB = W3t + ((n0 + r) << 7) + (hi << 3);   // col n0+r, K base hi*8
  v8f acc = {0.f,0.f,0.f,0.f,0.f,0.f,0.f,0.f};
#pragma unroll
  for (int kc = 0; kc < HID; kc += 32) {
    v16h a = cat8(*(const v8h*)(pA + kc), *(const v8h*)(pA + kc + 16));
    v16h b = cat8(*(const v8h*)(pB + kc), *(const v8h*)(pB + kc + 16));
    acc = __builtin_amdgcn_wmma_f32_16x16x32_f16(false, a, false, b, (short)0,
                                                 acc, false, false);
  }
  const int co  = n0 >> 6;
  const int ci  = (n0 & 63) + r;                       // D: col N = lane&15
#pragma unroll
  for (int vi = 0; vi < 8; ++vi) {
    const int j = j0 + vi + (hi << 3);                 // D: row M = vgpr + 8*hi
    Akern[(((co << 11) + j) << 6) + ci] = (_Float16)acc[vi];
  }
}

// ---- K3: causal conv as implicit GEMM with split-K over j, f32 atomics ----
// WG tile: 64 co x 64 t, j-chunk of 256. 512 threads = 16 waves, 1 subtile each.
__global__ void __launch_bounds__(512) ck_conv(const _Float16* __restrict__ Akern,
                                               const _Float16* __restrict__ xh,
                                               float* __restrict__ out) {
  const int jc = blockIdx.x;           // 0..7   (j-chunk)
  const int tt = blockIdx.y;           // 0..31  (t-tile)
  const int b  = blockIdx.z;           // 0..1
  const int t0 = tt << 6;
  const int j0 = jc << 8;
  const int sb = t0 + 1 + j0 - LEN;                    // = t0 + j0 - 2047
  if (sb + 318 < 0) return;                            // entire tile hits zero-pad region

  __shared__ _Float16 xlds[320 * 64];                  // x band: [srel][ci], 40 KB
  const int tid = threadIdx.x;

  // rows [0, nzrow): causal zero-pad -> ds_store zero fill
  const int nzrow = (sb < 0) ? -sb : 0;                // early-exit guarantees nzrow <= 318
  for (int cid = tid; cid < nzrow * 8; cid += 512) {
    const int row = cid >> 3, part = cid & 7;
    v8h v = {(_Float16)0,(_Float16)0,(_Float16)0,(_Float16)0,
             (_Float16)0,(_Float16)0,(_Float16)0,(_Float16)0};
    *(v8h*)(xlds + (row << 6) + (part << 3)) = v;
  }

  // rows [nzrow, 319): async DMA straight into LDS (no VGPR round-trip),
  // tracked with ASYNCcnt. 16B (b128) per lane per op, contiguous 128B rows.
  {
    // byte address of band row 0 in xh (may point before xh when sb<0; only
    // offsets >= nzrow*128 are issued, which land inside the array)
    const unsigned long long gbase =
        (unsigned long long)(uintptr_t)xh +
        (((long long)((b << 11) + sb)) << 7);          // *64 halves *2 bytes
    for (int cid = (nzrow << 3) + tid; cid < 319 * 8; cid += 512) {
      const unsigned goff = (unsigned)cid << 4;        // row*128 + part*16 bytes
      const unsigned lds  = (unsigned)(uintptr_t)xlds + goff;
      asm volatile("global_load_async_to_lds_b128 %0, %1, %2"
                   :: "v"(lds), "v"(goff), "s"(gbase)
                   : "memory");
    }
  }
  WAIT_ASYNC0();
  __syncthreads();

  const int wave = tid >> 5, lane = tid & 31;
  const int r = lane & 15, hi = lane >> 4;
  const int mi = wave & 3, ni = wave >> 2;             // co-subtile, t-subtile
  const int co0 = mi << 4;

  // A: row co0+r of Akern, at (j, ci-chunk); per-lane K base = hi*8 within chunk
  const _Float16* pA = Akern + ((((co0 + r) << 11) + j0) << 6) + (hi << 3);
  // B: column t = t0 + ni*16 + r  ->  srel = ni*16 + r + jt; per-lane K base = hi*8
  const _Float16* pB = xlds + (((ni << 4) + r) << 6) + (hi << 3);

  v8f acc0 = {0.f,0.f,0.f,0.f,0.f,0.f,0.f,0.f};
  v8f acc1 = {0.f,0.f,0.f,0.f,0.f,0.f,0.f,0.f};
#pragma unroll 2
  for (int jt = 0; jt < 256; ++jt) {
    v16h a0 = cat8(*(const v8h*)(pA +  0), *(const v8h*)(pA + 16));   // ci 0..31
    v16h a1 = cat8(*(const v8h*)(pA + 32), *(const v8h*)(pA + 48));   // ci 32..63
    v16h b0 = cat8(*(const v8h*)(pB +  0), *(const v8h*)(pB + 16));
    v16h b1 = cat8(*(const v8h*)(pB + 32), *(const v8h*)(pB + 48));
    acc0 = __builtin_amdgcn_wmma_f32_16x16x32_f16(false, a0, false, b0, (short)0,
                                                  acc0, false, false);
    acc1 = __builtin_amdgcn_wmma_f32_16x16x32_f16(false, a1, false, b1, (short)0,
                                                  acc1, false, false);
    pA += 64;                                          // next j row in Akern
    pB += 64;                                          // next srel row in band
  }

  const int tcol = t0 + (ni << 4) + r;                 // D: col N = lane&15
#pragma unroll
  for (int vi = 0; vi < 8; ++vi) {
    const int co = co0 + vi + (hi << 3);               // D: row M = vgpr + 8*hi
    atomicAdd(out + ((((b << 6) + co) << 11) + tcol), acc0[vi] + acc1[vi]);
  }
}

extern "C" void kernel_launch(void* const* d_in, const int* in_sizes, int n_in,
                              void* d_out, int out_size, void* d_ws, size_t ws_size,
                              hipStream_t stream) {
  (void)in_sizes; (void)n_in; (void)ws_size;
  const float* x  = (const float*)d_in[0];   // (2,64,2048)
  const float* W1 = (const float*)d_in[1];   // (1,128)
  const float* W2 = (const float*)d_in[2];   // (128,128)
  const float* W3 = (const float*)d_in[3];   // (128,4096)
  float* out = (float*)d_out;                // (2,64,2048)

  // workspace layout (f16): xh[262144] | h2h[262144] | W3t[524288] | Akern[8388608]  = 18 MB
  _Float16* xh    = (_Float16*)d_ws;
  _Float16* h2h   = xh  + (NB * LEN * CIN);
  _Float16* W3t   = h2h + (LEN * HID);
  _Float16* Akern = W3t + (COUT * CIN * HID);

  hipMemsetAsync(d_out, 0, (size_t)out_size * sizeof(float), stream);
  ck_cvt_x  <<<1024, 256, 0, stream>>>(x, xh);
  ck_cvt_w3 <<<2048, 256, 0, stream>>>(W3, W3t);
  ck_siren  <<<LEN, HID, 0, stream>>>(W1, W2, h2h);
  ck_gen_gemm<<<4096, 256, 0, stream>>>(h2h, W3t, Akern);
  ck_conv   <<<dim3(8, 32, 2), 512, 0, stream>>>(Akern, xh, out);
}
